// CausalSelfAttention_35158602285627
// MI455X (gfx1250) — matrix operand order repaired and checked by reading, not compiled, not run on hardware
//
#include <hip/hip_runtime.h>

// CDNA5 / gfx1250: wave32, bf16 WMMA 16x16x32 (f32 accum), async global->LDS staging.
typedef __attribute__((ext_vector_type(16))) __bf16 v16bf;
typedef __attribute__((ext_vector_type(8)))  float  v8f;
typedef int v4i_t __attribute__((vector_size(16)));   // matches builtin's V4i param

#define Bc  2
#define Tc  2048
#define Cc  1024
#define NHc 16
#define HDc 64
#define Mc  (Bc * Tc)   // 4096 rows of x / q / k / v / y

__device__ __forceinline__ unsigned short f2bf(float f) {
    unsigned int u = __float_as_uint(f);
    u += 0x7FFFu + ((u >> 16) & 1u);          // round-to-nearest-even
    return (unsigned short)(u >> 16);
}

union AFrag { v16bf v; unsigned int u[8]; };

// ---- async global->LDS copy (16B per lane), fallback = synchronous copy ----
#if defined(__has_builtin) && __has_builtin(__builtin_amdgcn_global_load_async_to_lds_b128)
#define HAVE_ASYNC_LDS 1
#else
#define HAVE_ASYNC_LDS 0
#endif

__device__ __forceinline__ void cp16_g2l(const unsigned short* g, unsigned short* l) {
#if HAVE_ASYNC_LDS
    __builtin_amdgcn_global_load_async_to_lds_b128(
        (__attribute__((address_space(1))) v4i_t*)g,
        (__attribute__((address_space(3))) v4i_t*)l, 0, 0);
#else
    *(uint4*)l = *(const uint4*)g;
#endif
}

template <int N>
__device__ __forceinline__ void wait_async() {
#if HAVE_ASYNC_LDS
#if defined(__has_builtin) && __has_builtin(__builtin_amdgcn_s_wait_asynccnt)
    __builtin_amdgcn_s_wait_asynccnt(N);
#else
    __asm__ volatile("s_wait_asynccnt %0" :: "i"(N) : "memory");
#endif
#endif
}

// ---------------------------------------------------------------- convert
__global__ void cvt_f32_bf16(const float* __restrict__ in,
                             unsigned short* __restrict__ out, int n) {
    int i = blockIdx.x * blockDim.x + threadIdx.x;
    if (i < n) out[i] = f2bf(in[i]);
}

// ---------------------------------------------------------------- GEMM  Y = X @ W^T
// X: [M,K] bf16 row-major.  W: [N,K] bf16 row-major (torch [out,in]).
// Block tile 128(M) x 128(N), 8 waves (4x2), each wave 32x64 = 2x4 WMMA tiles.
// Double-buffered LDS with async global->LDS copies (ASYNCcnt pipelining).
template <bool OUT_BF16>
__global__ __launch_bounds__(256) void gemm_xwt(
        const unsigned short* __restrict__ X,
        const unsigned short* __restrict__ W,
        void* __restrict__ outp, int M, int N, int Kd) {
    constexpr int LDT = 40;                       // padded stride (halves): 80B rows, 16B aligned
    __shared__ unsigned short As[2][128 * LDT];
    __shared__ unsigned short Bs[2][128 * LDT];

    const int tid  = threadIdx.x;
    const int lane = tid & 31, wave = tid >> 5;
    const int l    = lane & 15, hi = lane >> 4;
    const int wm   = wave & 3,  wn = wave >> 2;
    const int rowBase = blockIdx.x * 128;
    const int colBase = blockIdx.y * 128;

    // per-thread staging assignment: 2 A chunks + 2 B chunks of 16B (128 rows x 4 chunks each)
    auto stage = [&](unsigned short* Ad, unsigned short* Bd, int k0) {
        #pragma unroll
        for (int it = 0; it < 2; ++it) {
            const int ci = tid + it * 256;
            const int r = ci >> 2, c = ci & 3;
            cp16_g2l(&X[(size_t)(rowBase + r) * Kd + k0 + c * 8], &Ad[r * LDT + c * 8]);
            cp16_g2l(&W[(size_t)(colBase + r) * Kd + k0 + c * 8], &Bd[r * LDT + c * 8]);
        }
    };

    v8f acc[2][4] = {};
    const int nIter = Kd / 32;

    stage(As[0], Bs[0], 0);                       // prologue: tile 0 -> buffer 0

    for (int t = 0; t < nIter; ++t) {
        const int cur = t & 1;
        __syncthreads();                          // everyone done reading buf[cur^1]
        if (t + 1 < nIter) {
            stage(As[cur ^ 1], Bs[cur ^ 1], (t + 1) * 32);
            wait_async<4>();                      // tile t's 4 copies done (in-order), t+1 in flight
        } else {
            wait_async<0>();
        }
        __syncthreads();                          // all waves' current-tile data landed

        AFrag a[2], b[4];
        #pragma unroll
        for (int ti = 0; ti < 2; ++ti) {          // A frag 16x32: lane = row M
            const unsigned short* ab = &As[cur][(wm * 32 + ti * 16 + l) * LDT];
            #pragma unroll
            for (int v = 0; v < 4; ++v) a[ti].u[v] = *(const unsigned int*)&ab[2 * v + hi * 8];
            #pragma unroll
            for (int v = 4; v < 8; ++v) a[ti].u[v] = *(const unsigned int*)&ab[16 + 2 * (v - 4) + hi * 8];
        }
        #pragma unroll
        for (int tj = 0; tj < 4; ++tj) {          // B frag 32x16: lane = col N
            const unsigned short* bb = &Bs[cur][(wn * 64 + tj * 16 + l) * LDT];
            #pragma unroll
            for (int v = 0; v < 8; ++v) b[tj].u[v] = *(const unsigned int*)&bb[2 * (v & 3) + (v >> 2) * 16 + hi * 8];
        }
        #pragma unroll
        for (int ti = 0; ti < 2; ++ti)
            #pragma unroll
            for (int tj = 0; tj < 4; ++tj)
                acc[ti][tj] = __builtin_amdgcn_wmma_f32_16x16x32_bf16(
                    false, a[ti].v, false, b[tj].v, (short)0, acc[ti][tj], false, false);
    }

    // C/D layout: m = v + 8*hi, n = lane&15
    #pragma unroll
    for (int ti = 0; ti < 2; ++ti)
        #pragma unroll
        for (int tj = 0; tj < 4; ++tj)
            #pragma unroll
            for (int v = 0; v < 8; ++v) {
                const int m = rowBase + wm * 32 + ti * 16 + v + 8 * hi;
                const int n = colBase + wn * 64 + tj * 16 + l;
                if (OUT_BF16)
                    ((unsigned short*)outp)[(size_t)m * N + n] = f2bf(acc[ti][tj][v]);
                else
                    ((float*)outp)[(size_t)m * N + n] = acc[ti][tj][v];
            }
}

// ---------------------------------------------------------------- flash attention
// grid.x = B*NH (head), grid.y = T/128 (query block). 8 waves; each wave owns 16 q rows.
__global__ __launch_bounds__(256) void attn_fwd(
        const unsigned short* __restrict__ Qg,
        const unsigned short* __restrict__ Kg,
        const unsigned short* __restrict__ Vg,
        unsigned short* __restrict__ Yg) {
    constexpr int LDP = 40;
    __shared__ unsigned short Ps[8][16 * LDP];    // per-wave P bounce buffer

    const int tid  = threadIdx.x;
    const int lane = tid & 31, wave = tid >> 5;
    const int l    = lane & 15, hi = lane >> 4;
    const int bh = blockIdx.x;
    const int b  = bh / NHc, h = bh % NHc;
    const int qr0 = blockIdx.y * 128 + wave * 16;

    const size_t headOff = (size_t)b * Tc * Cc + (size_t)h * HDc;
    const unsigned short* Qh = Qg + headOff;
    const unsigned short* Kh = Kg + headOff;
    const unsigned short* Vh = Vg + headOff;
    unsigned short*       Yh = Yg + headOff;

    // Q A-fragments (16 rows x 64 dims => two 16x32 frags), loaded once
    AFrag qa[2];
    #pragma unroll
    for (int kc = 0; kc < 2; ++kc) {
        const unsigned short* qb = Qh + (size_t)(qr0 + l) * Cc + kc * 32;
        #pragma unroll
        for (int v = 0; v < 4; ++v) qa[kc].u[v] = *(const unsigned int*)&qb[2 * v + hi * 8];
        #pragma unroll
        for (int v = 4; v < 8; ++v) qa[kc].u[v] = *(const unsigned int*)&qb[16 + 2 * (v - 4) + hi * 8];
    }

    v8f o[4] = {};                                 // O accumulator 16x64
    float mrow[8], lrow[8];
    #pragma unroll
    for (int v = 0; v < 8; ++v) { mrow[v] = -1e30f; lrow[v] = 0.0f; }

    const int ntiles = (qr0 + 47) >> 5;            // causal trip count (32-key tiles)
    for (int tile = 0; tile < ntiles; ++tile) {
        const int j0 = tile * 32;

        // ---- S = Q @ K^T  (two 16x16 key tiles)
        v8f s[2] = {};
        #pragma unroll
        for (int nt = 0; nt < 2; ++nt) {
            const unsigned short* kb = Kh + (size_t)(j0 + nt * 16 + l) * Cc;
            #pragma unroll
            for (int kc = 0; kc < 2; ++kc) {
                AFrag kf;                          // K rows row-major == B-frag layout
                #pragma unroll
                for (int v = 0; v < 8; ++v)
                    kf.u[v] = *(const unsigned int*)&kb[kc * 32 + 2 * (v & 3) + (v >> 2) * 16 + hi * 8];
                s[nt] = __builtin_amdgcn_wmma_f32_16x16x32_bf16(
                    false, qa[kc].v, false, kf.v, (short)0, s[nt], false, false);
            }
        }

        // ---- scale + causal mask
        float p[2][8];
        const bool needMask = (j0 + 31) > qr0;
        #pragma unroll
        for (int nt = 0; nt < 2; ++nt)
            #pragma unroll
            for (int v = 0; v < 8; ++v) {
                float sv = s[nt][v] * 0.125f;      // 1/sqrt(64)
                if (needMask && (j0 + nt * 16 + l) > (qr0 + v + 8 * hi)) sv = -1e30f;
                p[nt][v] = sv;
            }

        // ---- online softmax (row reductions across the 16 lanes of each half)
        #pragma unroll
        for (int v = 0; v < 8; ++v) {
            float mv = fmaxf(p[0][v], p[1][v]);
            #pragma unroll
            for (int off = 1; off < 16; off <<= 1) mv = fmaxf(mv, __shfl_xor(mv, off, 32));
            float mn = fmaxf(mrow[v], mv);
            float e0 = __expf(p[0][v] - mn);
            float e1 = __expf(p[1][v] - mn);
            p[0][v] = e0; p[1][v] = e1;
            float sum = e0 + e1;
            #pragma unroll
            for (int off = 1; off < 16; off <<= 1) sum += __shfl_xor(sum, off, 32);
            float alpha = __expf(mrow[v] - mn);
            lrow[v] = lrow[v] * alpha + sum;
            mrow[v] = mn;
            #pragma unroll
            for (int dt = 0; dt < 4; ++dt) o[dt][v] *= alpha;   // rescale O
        }

        // ---- P (C-layout f32) -> LDS -> A-layout bf16 fragment
        unsigned short* pw = &Ps[wave][0];
        #pragma unroll
        for (int nt = 0; nt < 2; ++nt)
            #pragma unroll
            for (int v = 0; v < 8; ++v)
                pw[(v + 8 * hi) * LDP + nt * 16 + l] = f2bf(p[nt][v]);
        __asm__ volatile("s_wait_dscnt 0" ::: "memory");
        AFrag pa;
        {
            const unsigned short* pb = &pw[l * LDP];
            #pragma unroll
            for (int v = 0; v < 4; ++v) pa.u[v] = *(const unsigned int*)&pb[2 * v + hi * 8];
            #pragma unroll
            for (int v = 4; v < 8; ++v) pa.u[v] = *(const unsigned int*)&pb[16 + 2 * (v - 4) + hi * 8];
        }

        // ---- O += P @ V  (four 16-dim output tiles; V B-frags from paired u16 loads)
        #pragma unroll
        for (int dt = 0; dt < 4; ++dt) {
            AFrag vf;
            #pragma unroll
            for (int v = 0; v < 8; ++v) {
                const size_t kk = (size_t)(j0 + 2 * (v & 3) + (v >> 2) * 16 + hi * 8);
                unsigned int lo  = Vh[kk * Cc + dt * 16 + l];
                unsigned int hi2 = Vh[(kk + 1) * Cc + dt * 16 + l];
                vf.u[v] = lo | (hi2 << 16);
            }
            o[dt] = __builtin_amdgcn_wmma_f32_16x16x32_bf16(
                false, pa.v, false, vf.v, (short)0, o[dt], false, false);
        }
    }

    // ---- normalize and store Y (bf16, [B,T,H*D] layout => feeds final GEMM directly)
    #pragma unroll
    for (int v = 0; v < 8; ++v) {
        float inv = 1.0f / lrow[v];
        #pragma unroll
        for (int dt = 0; dt < 4; ++dt)
            Yh[(size_t)(qr0 + v + 8 * hi) * Cc + dt * 16 + l] = f2bf(o[dt][v] * inv);
    }
}

// ---------------------------------------------------------------- launch
extern "C" void kernel_launch(void* const* d_in, const int* in_sizes, int n_in,
                              void* d_out, int out_size, void* d_ws, size_t ws_size,
                              hipStream_t stream) {
    (void)in_sizes; (void)n_in; (void)out_size; (void)ws_size;
    const float* x  = (const float*)d_in[0];
    const float* Wq = (const float*)d_in[1];
    const float* Wk = (const float*)d_in[2];
    const float* Wv = (const float*)d_in[3];
    const float* Wo = (const float*)d_in[4];
    float* out = (float*)d_out;

    // workspace layout (bf16 halves): xb, 4 weights, Q, K, V, Y  => ~48 MB
    unsigned short* xb  = (unsigned short*)d_ws;
    unsigned short* wqb = xb  + (size_t)Mc * Cc;
    unsigned short* wkb = wqb + (size_t)Cc * Cc;
    unsigned short* wvb = wkb + (size_t)Cc * Cc;
    unsigned short* wob = wvb + (size_t)Cc * Cc;
    unsigned short* Qb  = wob + (size_t)Cc * Cc;
    unsigned short* Kb  = Qb  + (size_t)Mc * Cc;
    unsigned short* Vb  = Kb  + (size_t)Mc * Cc;
    unsigned short* Yb  = Vb  + (size_t)Mc * Cc;

    const int nx = Mc * Cc;      // 4M elems
    const int nw = Cc * Cc;      // 1M elems
    cvt_f32_bf16<<<(nx + 255) / 256, 256, 0, stream>>>(x,  xb,  nx);
    cvt_f32_bf16<<<(nw + 255) / 256, 256, 0, stream>>>(Wq, wqb, nw);
    cvt_f32_bf16<<<(nw + 255) / 256, 256, 0, stream>>>(Wk, wkb, nw);
    cvt_f32_bf16<<<(nw + 255) / 256, 256, 0, stream>>>(Wv, wvb, nw);
    cvt_f32_bf16<<<(nw + 255) / 256, 256, 0, stream>>>(Wo, wob, nw);

    dim3 gg(Mc / 128, Cc / 128);  // 32 x 8 blocks
    gemm_xwt<true><<<gg, 256, 0, stream>>>(xb, wqb, Qb, Mc, Cc, Cc);
    gemm_xwt<true><<<gg, 256, 0, stream>>>(xb, wkb, Kb, Mc, Cc, Cc);
    gemm_xwt<true><<<gg, 256, 0, stream>>>(xb, wvb, Vb, Mc, Cc, Cc);

    dim3 ga(Bc * NHc, Tc / 128); // 32 x 16 blocks
    attn_fwd<<<ga, 256, 0, stream>>>(Qb, Kb, Vb, Yb);

    gemm_xwt<false><<<gg, 256, 0, stream>>>(Yb, wob, out, Mc, Cc, Cc);
}
